// MetaLearner_3702261809702
// MI455X (gfx1250) — matrix operand order repaired and checked by reading, not compiled, not run on hardware
//
#include <hip/hip_runtime.h>
#include <hip/hip_bf16.h>

typedef __attribute__((ext_vector_type(16))) _Float16 v16h;
typedef __attribute__((ext_vector_type(8)))  float    v8f;
typedef __attribute__((ext_vector_type(4)))  int      v4i;

#define AS1 __attribute__((address_space(1)))
#define AS3 __attribute__((address_space(3)))

#define T_N 16
#define B_N 64
#define D_N 512
#define C_N 256
#define P_N 131072

// ---------------- workspace layout (bytes) ----------------
// h stored as WMMA A-fragment f16 image: [tile(8192)][kchunk(2)][lane(32)][8 x u32]
#define OFF_HFRAG 0ull
#define SZ_HFRAG  (8192ull*2*32*8*4)          // 16 MB
// c stored as WMMA C-accumulator f32 image: [tile][j(3)][lane(32)][8 x f32]
#define OFF_CFRAG (OFF_HFRAG + SZ_HFRAG)
#define SZ_CFRAG  (8192ull*3*32*8*4)          // 24 MB
#define OFF_P     (OFF_CFRAG + SZ_CFRAG)      // f32 [P]
#define OFF_PH    (OFF_P  + 4ull*P_N)         // f16 [P]
#define OFF_F     (OFF_PH + 2ull*P_N)         // f32 [P]
#define OFF_I     (OFF_F  + 4ull*P_N)         // f32 [P]
#define OFF_G     (OFF_I  + 4ull*P_N)         // f32 [P] grads
#define OFF_E     (OFF_G  + 4ull*P_N)         // f16 [64][256]
#define OFF_LOSS  (OFF_E  + 2ull*B_N*C_N)     // f32 [16]
#define OFF_WZ    (OFF_LOSS + 64ull)          // f16 frag image: [g(4)][kc(2)][nt(12)][lane(32)][16]
#define OFF_BIAS  (OFF_WZ + 4ull*12288*2)     // f32 [4][192]

union AFrag { v16h h; unsigned u[8]; uint4 q[2]; };

__device__ inline unsigned pack2h(float a, float b){
  union { _Float16 h[2]; unsigned u; } x;
  x.h[0] = (_Float16)a; x.h[1] = (_Float16)b; return x.u;
}
__device__ inline unsigned packhh(_Float16 a, _Float16 b){
  union { _Float16 h[2]; unsigned u; } x;
  x.h[0] = a; x.h[1] = b; return x.u;
}
__device__ inline float hsig(float x){ return fminf(fmaxf(0.2f*x + 0.5f, 0.f), 1.f); }

// ---------------- init: zero state, copy params ----------------
__global__ void k_init(const float* __restrict__ params, char* __restrict__ ws){
  size_t stride = (size_t)gridDim.x * blockDim.x;
  size_t i0 = (size_t)blockIdx.x * blockDim.x + threadIdx.x;
  float* zf = (float*)(ws + OFF_HFRAG);
  size_t zwords = (SZ_HFRAG + SZ_CFRAG) / 4;
  for (size_t i = i0; i < zwords; i += stride) zf[i] = 0.f;
  float*    p  = (float*)(ws + OFF_P);
  _Float16* ph = (_Float16*)(ws + OFF_PH);
  float*    fb = (float*)(ws + OFF_F);
  float*    ib = (float*)(ws + OFF_I);
  for (size_t i = i0; i < P_N; i += stride){
    float v = params[i];
    p[i] = v; ph[i] = (_Float16)v; fb[i] = 0.f; ib[i] = 0.f;
  }
  if (i0 < 16) ((float*)(ws + OFF_LOSS))[i0] = 0.f;
}

// ---------------- prep: build padded/fused weight frags ----------------
// Combined input order along K: [h(0..39) | p,g0,g1,l0,l1 (40..44) | pad..63]
// Combined output cols: gate*48 + u (u<40 valid), 192 total.
__global__ void k_prepw(const float* __restrict__ kern, const float* __restrict__ rk,
                        const float* __restrict__ bias, char* __restrict__ ws){
  int g = blockIdx.x;
  _Float16* wzf = (_Float16*)(ws + OFF_WZ) + (size_t)g * 12288;
  for (int s = threadIdx.x; s < 12288; s += blockDim.x){
    int kc = s / 6144; int r = s % 6144;
    int nt = r / 512;  int r2 = r % 512;
    int lane = r2 / 16; int e = r2 % 16;
    int K = kc*32 + ((e < 8) ? ((lane >> 4)*8 + e) : (16 + (lane >> 4)*8 + (e - 8)));
    int N = nt*16 + (lane & 15);
    int gate = N / 48, u = N % 48;
    float x = 0.f;
    if (u < 40){
      int co = gate*40 + u;
      if (K < 40)      x = rk[((size_t)g*40 + K)*160 + co];
      else if (K < 45) x = kern[((size_t)g*5 + (K - 40))*160 + co];
    }
    wzf[s] = (_Float16)x;
  }
  float* bp = (float*)(ws + OFF_BIAS) + g*192;
  for (int s = threadIdx.x; s < 192; s += blockDim.x){
    int gate = s / 48, u = s % 48;
    bp[s] = (u < 40) ? bias[(size_t)g*160 + gate*40 + u] : 0.f;
  }
}

// ---------------- step phase 1: E = 2(pred-y)/BC, loss ----------------
__global__ __launch_bounds__(256) void k1_err(const float* __restrict__ feats,
    const float* __restrict__ labels, char* __restrict__ ws, int t){
  int tid = threadIdx.x;
  int wid = blockIdx.x*8 + (tid >> 5);
  int lane = tid & 31, n = lane & 15, hi = lane >> 4;
  int tb = wid >> 4, tc = wid & 15;
  const float* f_t = feats  + (size_t)t*B_N*D_N;
  const float* y_t = labels + (size_t)t*B_N*C_N;
  const _Float16* ph = (const _Float16*)(ws + OFF_PH);
  v8f acc;
  #pragma unroll
  for (int v = 0; v < 8; ++v) acc[v] = 0.f;
  int arow = tb*16 + n;        // batch row for this lane's A rows
  int ncol = tc*16 + n;        // output column for this lane
  for (int kc = 0; kc < 16; ++kc){
    AFrag A, Bm;
    #pragma unroll
    for (int v = 0; v < 8; ++v){
      int kb = kc*32 + hi*16 + 2*v;
      A.u[v] = pack2h(f_t[(size_t)arow*D_N + kb], f_t[(size_t)arow*D_N + kb + 1]);
    }
    #pragma unroll
    for (int e = 0; e < 16; e += 2){
      int K = kc*32 + hi*16 + e;
      Bm.u[e >> 1] = packhh(ph[(size_t)K*C_N + ncol], ph[(size_t)(K + 1)*C_N + ncol]);
    }
    acc = __builtin_amdgcn_wmma_f32_16x16x32_f16(false, A.h, false, Bm.h,
                                                 (short)0, acc, false, false);
  }
  _Float16* E = (_Float16*)(ws + OFF_E);
  float* loss = (float*)(ws + OFF_LOSS);
  float lsum = 0.f;
  #pragma unroll
  for (int v = 0; v < 8; ++v){
    int m = tb*16 + v + 8*hi;
    float y = y_t[(size_t)m*C_N + ncol];
    float d = acc[v] - y;                       // pred - y
    E[(size_t)m*C_N + ncol] = (_Float16)(d * (2.f/16384.f));
    lsum += d*d;
  }
  atomicAdd(&loss[t], lsum * (1.f/16384.f));
}

// ---------------- step phase 2: grads = f^T @ E ----------------
__global__ __launch_bounds__(256) void k2_grads(const float* __restrict__ feats,
    char* __restrict__ ws, int t){
  int tid = threadIdx.x;
  int wid = blockIdx.x*8 + (tid >> 5);
  int lane = tid & 31, n = lane & 15, hi = lane >> 4;
  int td = wid >> 4, tc = wid & 15;
  const float* f_t = feats + (size_t)t*B_N*D_N;
  const _Float16* E = (const _Float16*)(ws + OFF_E);
  float* grads = (float*)(ws + OFF_G);
  int drow = td*16 + n;
  int ccol = tc*16 + n;
  v8f acc;
  #pragma unroll
  for (int v = 0; v < 8; ++v) acc[v] = 0.f;
  #pragma unroll
  for (int kc = 0; kc < 2; ++kc){
    AFrag A, Bm;
    #pragma unroll
    for (int v = 0; v < 8; ++v){
      int kb = kc*32 + hi*16 + 2*v;
      A.u[v] = pack2h(f_t[(size_t)kb*D_N + drow], f_t[(size_t)(kb + 1)*D_N + drow]);
    }
    #pragma unroll
    for (int e = 0; e < 16; e += 2){
      int K = kc*32 + hi*16 + e;
      Bm.u[e >> 1] = packhh(E[(size_t)K*C_N + ccol], E[(size_t)(K + 1)*C_N + ccol]);
    }
    acc = __builtin_amdgcn_wmma_f32_16x16x32_f16(false, A.h, false, Bm.h,
                                                 (short)0, acc, false, false);
  }
  #pragma unroll
  for (int v = 0; v < 8; ++v)
    grads[(size_t)(td*16 + v + 8*hi)*C_N + ccol] = acc[v];
}

// ---------------- step phase 3: coordinate-wise LSTM via WMMA ----------------
__global__ __launch_bounds__(256) void k3_lstm(
    const float* __restrict__ W_f, const float* __restrict__ b_f,
    const float* __restrict__ W_i, const float* __restrict__ b_i,
    char* __restrict__ ws, int t){
  __shared__ _Float16 wz_lds[2*12*32*16];   // 24576 B: B-frag image for this group
  __shared__ _Float16 h_lds[8][16][48];     // 12288 B: per-wave h transpose buffer

  int tid = threadIdx.x, wave = tid >> 5, lane = tid & 31;
  int n = lane & 15, hi = lane >> 4;
  int coordBase = blockIdx.x * 128;
  int g = (coordBase < 16384) ? 0 : (coordBase < 49152) ? 1 : (coordBase < 98304) ? 2 : 3;

  int tileBase = coordBase + wave*16;
  size_t tIdx = (size_t)(tileBase >> 4);

  // L2->LDS latency hiding for operands consumed after the WMMA block
  __builtin_prefetch((const char*)ws + OFF_CFRAG + tIdx*3072, 0, 0);
  __builtin_prefetch((const char*)ws + OFF_G + 4ull*(size_t)tileBase, 0, 0);

  { // stage group's weight frags into LDS (async direct-to-LDS when available)
    const uint4* src = (const uint4*)(ws + OFF_WZ) + (size_t)g * 1536;
    uint4* dst = (uint4*)wz_lds;
#if defined(__gfx1250__) && __has_builtin(__builtin_amdgcn_global_load_async_to_lds_b128)
    for (int i = tid; i < 1536; i += 256){
      __builtin_amdgcn_global_load_async_to_lds_b128(
          (AS1 v4i*)(src + i), (AS3 v4i*)(dst + i), 0, 0);
    }
  #if __has_builtin(__builtin_amdgcn_s_wait_asynccnt)
    __builtin_amdgcn_s_wait_asynccnt(0);
  #else
    asm volatile("s_wait_asynccnt 0x0" ::: "memory");
  #endif
#else
    for (int i = tid; i < 1536; i += 256) dst[i] = src[i];
#endif
  }
  __syncthreads();

  unsigned* hfr = (unsigned*)(ws + OFF_HFRAG) + tIdx*512 + (size_t)lane*8;

  AFrag a0, a1;
  a0.q[0] = ((const uint4*)hfr)[0];         a0.q[1] = ((const uint4*)hfr)[1];
  a1.q[0] = ((const uint4*)(hfr + 256))[0]; a1.q[1] = ((const uint4*)(hfr + 256))[1];

  float*    pbuf  = (float*)(ws + OFF_P);
  _Float16* phbuf = (_Float16*)(ws + OFF_PH);
  float*    fbuf  = (float*)(ws + OFF_F);
  float*    ibuf  = (float*)(ws + OFF_I);
  const float* gbuf = (const float*)(ws + OFF_G);

  int coord = tileBase + n;                // this lane's coordinate row
  float pm = pbuf[coord];
  float gm = gbuf[coord];
  float Lv = ((const float*)(ws + OFF_LOSS))[t];

  // preprocess (Andrychowicz et al., PREP=10)
  const float NEP = 4.5399929762484854e-05f;   // exp(-10)
  const float EP  = 22026.465794806718f;       // exp(10)
  float ag = fabsf(gm); int m1g = ag > NEP;
  float g0 = m1g ? (logf(ag) * 0.1f) : -1.f;
  float g1 = m1g ? ((gm > 0.f) ? 1.f : -1.f) : (EP * gm);
  float al = fabsf(Lv); int m1l = al > NEP;
  float l0 = m1l ? (logf(al) * 0.1f) : -1.f;
  float l1 = m1l ? ((Lv > 0.f) ? 1.f : -1.f) : (EP * Lv);

  // patch scalar inputs (K=40..44) into A chunk-1 regs of hi-lanes (no divergence)
  a1.u[0] = hi ? pack2h(pm, g0) : a1.u[0];
  a1.u[1] = hi ? pack2h(g1, l0) : a1.u[1];
  a1.u[2] = hi ? pack2h(l1, 0.f) : a1.u[2];

  const float* biasp = (const float*)(ws + OFF_BIAS) + g*192;
  v8f acc[12];
  #pragma unroll
  for (int nt = 0; nt < 12; ++nt){
    float bv = biasp[nt*16 + n];
    v8f ci;
    #pragma unroll
    for (int v = 0; v < 8; ++v) ci[v] = bv;
    AFrag b0, b1;
    const uint4* w0 = (const uint4*)(wz_lds + (size_t)nt*512 + (size_t)lane*16);
    const uint4* w1 = (const uint4*)(wz_lds + (size_t)(12 + nt)*512 + (size_t)lane*16);
    b0.q[0] = w0[0]; b0.q[1] = w0[1];
    b1.q[0] = w1[0]; b1.q[1] = w1[1];
    v8f z = __builtin_amdgcn_wmma_f32_16x16x32_f16(false, a0.h, false, b0.h,
                                                   (short)0, ci, false, false);
    acc[nt] = __builtin_amdgcn_wmma_f32_16x16x32_f16(false, a1.h, false, b1.h,
                                                     (short)0, z, false, false);
  }

  // gates + cell update; c kept in accumulator-image layout
  float* cfr = (float*)(ws + OFF_CFRAG) + tIdx*768 + (size_t)lane*8;
  #pragma unroll
  for (int j = 0; j < 3; ++j){
    v8f cp = *(const v8f*)(cfr + j*256);
    v8f cn;
    #pragma unroll
    for (int v = 0; v < 8; ++v){
      float zi = acc[j][v], zf = acc[3 + j][v], zg = acc[6 + j][v], zo = acc[9 + j][v];
      float c2 = hsig(zf)*cp[v] + hsig(zi)*tanhf(zg);
      float h2 = hsig(zo)*tanhf(c2);
      cn[v] = c2;
      h_lds[wave][v + 8*hi][16*j + n] = (_Float16)h2;   // transpose via LDS
    }
    *(v8f*)(cfr + j*256) = cn;
  }

  // f_new / i_new: 40-dim dots split across lane pairs, reduced with shfl_xor(16)
  float df = 0.f, di = 0.f;
  const float* wfg = W_f + g*41;
  const float* wig = W_i + g*41;
  #pragma unroll
  for (int k = 0; k < 20; ++k){
    int u = hi*20 + k;
    float hv = (float)h_lds[wave][n][u];
    df += hv * wfg[u];
    di += hv * wig[u];
  }
  df += __shfl_xor(df, 16, 32);
  di += __shfl_xor(di, 16, 32);
  float fp = fbuf[coord], ipv = ibuf[coord];
  float fn = fmaxf(df + fp*wfg[40] + b_f[g], 0.f);
  float inw = fmaxf(di + ipv*wig[40] + b_i[g], 0.f);
  float pn = fn*pm - inw*gm;
  if (hi == 0){
    pbuf[coord] = pn; phbuf[coord] = (_Float16)pn;
    fbuf[coord] = fn; ibuf[coord] = inw;
  }

  // repack h (K=0..39) into A-fragment image for next step
  AFrag s0, s1;
  #pragma unroll
  for (int v = 0; v < 8; ++v){
    int K = (v < 4) ? (hi*8 + 2*v) : (16 + hi*8 + 2*(v - 4));
    s0.u[v] = packhh(h_lds[wave][n][K], h_lds[wave][n][K + 1]);
  }
  #pragma unroll
  for (int v = 0; v < 4; ++v){
    int K = 32 + 2*v;
    unsigned val = packhh(h_lds[wave][n][K], h_lds[wave][n][K + 1]);
    s1.u[v] = hi ? 0u : val;                 // hi-lanes' K>=40 slots stay zero
  }
  s1.u[4] = 0u; s1.u[5] = 0u; s1.u[6] = 0u; s1.u[7] = 0u;
  ((uint4*)hfr)[0] = s0.q[0];         ((uint4*)hfr)[1] = s0.q[1];
  ((uint4*)(hfr + 256))[0] = s1.q[0]; ((uint4*)(hfr + 256))[1] = s1.q[1];
}

// ---------------- host launch ----------------
extern "C" void kernel_launch(void* const* d_in, const int* in_sizes, int n_in,
                              void* d_out, int out_size, void* d_ws, size_t ws_size,
                              hipStream_t stream){
  (void)in_sizes; (void)n_in; (void)out_size; (void)ws_size;
  const float* feats  = (const float*)d_in[0];
  const float* labels = (const float*)d_in[1];
  const float* params = (const float*)d_in[2];
  const float* kern   = (const float*)d_in[3];
  const float* rk     = (const float*)d_in[4];
  const float* bias   = (const float*)d_in[5];
  const float* W_f    = (const float*)d_in[6];
  const float* b_f    = (const float*)d_in[7];
  const float* W_i    = (const float*)d_in[8];
  const float* b_i    = (const float*)d_in[9];
  char* ws = (char*)d_ws;

  k_init <<<2048, 256, 0, stream>>>(params, ws);
  k_prepw<<<4,    256, 0, stream>>>(kern, rk, bias, ws);
  for (int t = 0; t < T_N; ++t){
    k1_err  <<<8,    256, 0, stream>>>(feats, labels, ws, t);
    k2_grads<<<64,   256, 0, stream>>>(feats, ws, t);
    k3_lstm <<<1024, 256, 0, stream>>>(W_f, b_f, W_i, b_i, ws, t);
  }
  (void)hipMemcpyAsync(d_out, ws + OFF_P, sizeof(float)*P_N,
                       hipMemcpyDeviceToDevice, stream);
}
